// SelfAttention_7370163880718
// MI455X (gfx1250) — matrix operand order repaired and checked by reading, not compiled, not run on hardware
//
#include <hip/hip_runtime.h>
#include <hip/hip_bf16.h>

typedef __attribute__((ext_vector_type(16))) _Float16 v16h;
typedef __attribute__((ext_vector_type(8)))  _Float16 v8h;
typedef __attribute__((ext_vector_type(4)))  _Float16 v4h;
typedef __attribute__((ext_vector_type(2)))  _Float16 v2h;
typedef __attribute__((ext_vector_type(8)))  float    v8f;
typedef __attribute__((ext_vector_type(4)))  int      v4i;

#define BB    8
#define SS    4096
#define HH    8
#define HKV   2
#define DD    64
#define HID   512           // HH*DD
#define NQKV  768           // 512 (q) + 128 (k) + 128 (v)
#define MTOT  (BB*SS)       // 32768

union FragU { v16h v; v8h h[2]; };

// --- CDNA5 async global->LDS path (guarded; falls back to sync staging) ----
#if defined(__gfx1250__) && __has_builtin(__builtin_amdgcn_global_load_async_to_lds_b128)
#define HAS_ASYNC_LDS 1
#define AS1P(p) ((__attribute__((address_space(1))) v4i*)(p))
#define AS3P(p) ((__attribute__((address_space(3))) v4i*)(p))
#if __has_builtin(__builtin_amdgcn_s_wait_asynccnt)
#define ASYNC_WAIT(n) __builtin_amdgcn_s_wait_asynccnt(n)
#else
#define ASYNC_WAIT(n) asm volatile("s_wait_asynccnt %0" ::"i"(n) : "memory")
#endif
#else
#define HAS_ASYNC_LDS 0
#endif

// ---------------------------------------------------------------------------
// K1: cast fp32 weights to f16, transposed to (N x K) row-major so WMMA B
// fragments are contiguous-per-lane loads.
// wqkvT: 768 x 512 (cols 0..511 = wq, 512..639 = wk, 640..767 = wv)
// woT:   512 x 512
// ---------------------------------------------------------------------------
__global__ __launch_bounds__(256) void cast_weights_kernel(
    const float* __restrict__ wq, const float* __restrict__ wk,
    const float* __restrict__ wv, const float* __restrict__ wo,
    _Float16* __restrict__ wqkvT, _Float16* __restrict__ woT) {
  int idx = blockIdx.x * 256 + threadIdx.x;
  const int total1 = NQKV * HID;
  if (idx < total1) {
    int n = idx / HID, k = idx % HID;
    float val;
    if (n < 512)      val = wq[k * 512 + n];
    else if (n < 640) val = wk[k * 128 + (n - 512)];
    else              val = wv[k * 128 + (n - 640)];
    wqkvT[idx] = (_Float16)val;
  } else {
    int idx2 = idx - total1;
    if (idx2 < HID * HID) {
      int n = idx2 / HID, k = idx2 % HID;
      woT[idx2] = (_Float16)wo[k * HID + n];
    }
  }
}

// ---------------------------------------------------------------------------
// K2: cast x (fp32) -> xh (f16), 4 elements / thread
// ---------------------------------------------------------------------------
__global__ __launch_bounds__(256) void cast_x_kernel(
    const float* __restrict__ x, _Float16* __restrict__ xh, int n4) {
  int i = blockIdx.x * 256 + threadIdx.x;
  if (i < n4) {
    float4 v = ((const float4*)x)[i];
    v4h h;
    h.x = (_Float16)v.x; h.y = (_Float16)v.y;
    h.z = (_Float16)v.z; h.w = (_Float16)v.w;
    ((v4h*)xh)[i] = h;
  }
}

// ---------------------------------------------------------------------------
// K3/K5: WMMA GEMM  C(MxN) = A(MxK, f16 row-major) * Bt(NxK, f16 row-major)^T
// Block tile 128x128, 8 waves (4x2), each wave 32x64 = 2x4 WMMA 16x16 tiles.
// K staged through double-buffered LDS in 32-wide slices using the CDNA5
// async global->LDS DMA path (ASYNCcnt), one stage ahead of compute.
// ---------------------------------------------------------------------------
template <typename OutT>
__global__ __launch_bounds__(256) void gemm_wmma_kernel(
    const _Float16* __restrict__ A, const _Float16* __restrict__ Bt,
    OutT* __restrict__ C, int M, int N, int K) {
#if HAS_ASYNC_LDS
  __shared__ _Float16 lA[2][128][32];
  __shared__ _Float16 lB[2][128][32];
#else
  __shared__ _Float16 lA[1][128][32];
  __shared__ _Float16 lB[1][128][32];
#endif

  const int tid   = threadIdx.x;
  const int lane  = tid & 31;
  const int wave  = tid >> 5;
  const int wm    = wave >> 1;    // 0..3 (M direction)
  const int wn    = wave & 1;     // 0..1 (N direction)
  const int lhalf = lane >> 4;    // lane group (0: lanes 0-15, 1: 16-31)
  const int l16   = lane & 15;

  const int row0 = blockIdx.x * 128;
  const int col0 = blockIdx.y * 128;

  v8f acc[2][4] = {};

#if HAS_ASYNC_LDS
  // Issue one stage: 4 async 16B DMA copies per thread (2 for A, 2 for B).
  auto stage = [&](int kt, int buf) {
#pragma unroll
    for (int c = 0; c < 2; ++c) {
      int chunk = tid * 2 + c;              // 0..511
      int r  = chunk >> 2;                  // 0..127
      int k8 = (chunk & 3) * 8;             // 0,8,16,24
      __builtin_amdgcn_global_load_async_to_lds_b128(
          AS1P(&A[(size_t)(row0 + r) * K + kt + k8]), AS3P(&lA[buf][r][k8]),
          0, 0);
      __builtin_amdgcn_global_load_async_to_lds_b128(
          AS1P(&Bt[(size_t)(col0 + r) * K + kt + k8]), AS3P(&lB[buf][r][k8]),
          0, 0);
    }
  };

  stage(0, 0);
  const int nIt = K / 32;
  for (int it = 0; it < nIt; ++it) {
    const int buf = it & 1;
    if (it + 1 < nIt) {
      stage((it + 1) * 32, buf ^ 1);        // next stage DMA in flight
      ASYNC_WAIT(4);                        // current stage complete (in-order)
    } else {
      ASYNC_WAIT(0);
    }
    __syncthreads();                        // all waves' DMA visible
#else
  for (int it = 0; it < K / 32; ++it) {
    const int buf = 0;
#pragma unroll
    for (int c = 0; c < 2; ++c) {
      int chunk = tid * 2 + c;
      int r  = chunk >> 2;
      int k8 = (chunk & 3) * 8;
      *(v8h*)&lA[0][r][k8] = *(const v8h*)&A[(size_t)(row0 + r) * K + it * 32 + k8];
      *(v8h*)&lB[0][r][k8] = *(const v8h*)&Bt[(size_t)(col0 + r) * K + it * 32 + k8];
    }
    __syncthreads();
#endif

    // --- build fragments (ISA 7.12.2 layouts)
    FragU fa[2], fb[4];
#pragma unroll
    for (int tm = 0; tm < 2; ++tm) {
      int m = wm * 32 + tm * 16 + l16;
      fa[tm].h[0] = *(const v8h*)&lA[buf][m][lhalf * 8];      // K = lhalf*8..+7
      fa[tm].h[1] = *(const v8h*)&lA[buf][m][16 + lhalf * 8]; // K = 16+lhalf*8..
    }
#pragma unroll
    for (int tn = 0; tn < 4; ++tn) {
      int n = wn * 64 + tn * 16 + l16;
      fb[tn].h[0] = *(const v8h*)&lB[buf][n][lhalf * 8];      // K = lhalf*8..+7
      fb[tn].h[1] = *(const v8h*)&lB[buf][n][16 + lhalf * 8]; // K = 16+lhalf*8..
    }

#pragma unroll
    for (int tm = 0; tm < 2; ++tm)
#pragma unroll
      for (int tn = 0; tn < 4; ++tn)
        acc[tm][tn] = __builtin_amdgcn_wmma_f32_16x16x32_f16(
            false, fa[tm].v, false, fb[tn].v, (short)0, acc[tm][tn],
            false, false);
    __syncthreads();                        // buffer reuse guard
  }

  // --- epilogue: C/D layout: lane l16 = col, VGPR j = row j + 8*lhalf
#pragma unroll
  for (int tm = 0; tm < 2; ++tm) {
#pragma unroll
    for (int tn = 0; tn < 4; ++tn) {
      int col = col0 + wn * 64 + tn * 16 + l16;
#pragma unroll
      for (int j = 0; j < 8; ++j) {
        int row = row0 + wm * 32 + tm * 16 + j + 8 * lhalf;
        C[(size_t)row * N + col] = (OutT)acc[tm][tn][j];
      }
    }
  }
}

// ---------------------------------------------------------------------------
// K4: RoPE + cross-head causal attention, one wave32 per (b,s) position.
// Lane i owns dim pair (2i, 2i+1) -> RoPE is lane-local; dots over D=64
// reduce with 5 xor-shuffles. Output stored f16 in the permuted layout that
// matches reference transpose(0,2,1,3).reshape:
//   row = b*4096 + h*512 + s/8 ; col = (s%8)*64 + d
// ---------------------------------------------------------------------------
__global__ __launch_bounds__(256) void attn_rope_kernel(
    const _Float16* __restrict__ qkv, const float* __restrict__ fcos,
    const float* __restrict__ fsin, _Float16* __restrict__ ao) {
  const int tid  = threadIdx.x;
  const int lane = tid & 31;
  const int wave = tid >> 5;
  const int pos  = blockIdx.x * 8 + wave;   // 0..32767
  const int b    = pos >> 12;
  const int s    = pos & 4095;

  const _Float16* row = qkv + (size_t)pos * NQKV;
  const float c  = fcos[s * 32 + lane];
  const float sn = fsin[s * 32 + lane];

  float qe[8], qo[8], ke[2], ko[2], ve[2], vo[2];
#pragma unroll
  for (int h = 0; h < 8; ++h) {
    v2h p = *(const v2h*)&row[h * 64 + 2 * lane];
    float xr = (float)p.x, xi = (float)p.y;
    qe[h] = xr * c - xi * sn;
    qo[h] = xr * sn + xi * c;
  }
#pragma unroll
  for (int h = 0; h < 2; ++h) {
    v2h p = *(const v2h*)&row[512 + h * 64 + 2 * lane];
    float xr = (float)p.x, xi = (float)p.y;
    ke[h] = xr * c - xi * sn;
    ko[h] = xr * sn + xi * c;
    v2h pv = *(const v2h*)&row[640 + h * 64 + 2 * lane];
    ve[h] = (float)pv.x;
    vo[h] = (float)pv.y;
  }

  // scores, lower triangle only (causal over heads), GQA: kv head = kh>>2
  float sc[8][8];
  const float scale = 0.125f;   // 1/sqrt(64)
#pragma unroll
  for (int qh = 0; qh < 8; ++qh) {
#pragma unroll
    for (int kh = 0; kh <= qh; ++kh) {
      float p = qe[qh] * ke[kh >> 2] + qo[qh] * ko[kh >> 2];
      p += __shfl_xor(p, 16, 32);
      p += __shfl_xor(p, 8, 32);
      p += __shfl_xor(p, 4, 32);
      p += __shfl_xor(p, 2, 32);
      p += __shfl_xor(p, 1, 32);
      sc[qh][kh] = p * scale;
    }
  }

#pragma unroll
  for (int qh = 0; qh < 8; ++qh) {
    float m = sc[qh][0];
#pragma unroll
    for (int kh = 1; kh < 8; ++kh)
      if (kh <= qh) m = fmaxf(m, sc[qh][kh]);
    float w[8], sum = 0.f;
#pragma unroll
    for (int kh = 0; kh < 8; ++kh) {
      w[kh] = (kh <= qh) ? __expf(sc[qh][kh] - m) : 0.f;
      sum += w[kh];
    }
    float inv = 1.0f / sum;
    float oe = 0.f, oo = 0.f;
#pragma unroll
    for (int kh = 0; kh < 8; ++kh) {
      oe += w[kh] * ve[kh >> 2];
      oo += w[kh] * vo[kh >> 2];
    }
    oe *= inv; oo *= inv;

    size_t rp = (size_t)(b * 4096 + qh * 512 + (s >> 3)) * 512
              + (size_t)((s & 7) * 64 + 2 * lane);
    v2h o; o.x = (_Float16)oe; o.y = (_Float16)oo;
    *(v2h*)&ao[rp] = o;
  }
}

// ---------------------------------------------------------------------------
// Host-side orchestration
// ---------------------------------------------------------------------------
extern "C" void kernel_launch(void* const* d_in, const int* in_sizes, int n_in,
                              void* d_out, int out_size, void* d_ws,
                              size_t ws_size, hipStream_t stream) {
  const float* x    = (const float*)d_in[0];
  const float* fcos = (const float*)d_in[1];
  const float* fsin = (const float*)d_in[2];
  const float* wq   = (const float*)d_in[3];
  const float* wk   = (const float*)d_in[4];
  const float* wv   = (const float*)d_in[5];
  const float* wo   = (const float*)d_in[6];
  float* out = (float*)d_out;

  char* ws = (char*)d_ws;
  _Float16* xh    = (_Float16*)(ws);                 // 32768*512*2  = 33,554,432
  _Float16* wqkvT = (_Float16*)(ws + 33554432);      //   768*512*2  =    786,432
  _Float16* woT   = (_Float16*)(ws + 34340864);      //   512*512*2  =    524,288
  _Float16* qkv   = (_Float16*)(ws + 34865152);      // 32768*768*2  = 50,331,648
  _Float16* ao    = (_Float16*)(ws + 85196800);      // 32768*512*2  = 33,554,432
  // total workspace: 118,751,232 bytes

  // K1: weights -> f16 transposed
  {
    int total = NQKV * HID + HID * HID;              // 655,360
    cast_weights_kernel<<<(total + 255) / 256, 256, 0, stream>>>(
        wq, wk, wv, wo, wqkvT, woT);
  }
  // K2: x -> f16
  {
    int n4 = MTOT * HID / 4;                         // 4,194,304
    cast_x_kernel<<<n4 / 256, 256, 0, stream>>>(x, xh, n4);
  }
  // K3: QKV = XH @ WqkvT^T  (f16 out)
  {
    dim3 grid(MTOT / 128, NQKV / 128);               // 256 x 6
    gemm_wmma_kernel<_Float16><<<grid, 256, 0, stream>>>(
        xh, wqkvT, qkv, MTOT, NQKV, HID);
  }
  // K4: RoPE + attention, permuted f16 store
  {
    attn_rope_kernel<<<MTOT / 8, 256, 0, stream>>>(qkv, fcos, fsin, ao);
  }
  // K5: OUT = AO @ WoT^T  (f32 out)
  {
    dim3 grid(MTOT / 128, HID / 128);                // 256 x 4
    gemm_wmma_kernel<float><<<grid, 256, 0, stream>>>(
        ao, woT, out, MTOT, HID, HID);
  }
}
